// MultiAttentionBlock_15118284882512
// MI455X (gfx1250) — matrix operand order repaired and checked by reading, not compiled, not run on hardware
//
#include <hip/hip_runtime.h>
#include <hip/hip_bf16.h>

// Problem constants (match reference)
#define BB  4
#define TT  2048
#define CC  1024
#define HH  16
#define DHH 64

typedef __attribute__((ext_vector_type(16))) __bf16 v16bf;
typedef __attribute__((ext_vector_type(8)))  __bf16 v8bf;
typedef __attribute__((ext_vector_type(8)))  float  v8f;
typedef __attribute__((ext_vector_type(4)))  int    v4i_;

// ---------------------------------------------------------------------------
// Scheduler fence: force "issue all loads, then all WMMAs" so the register
// allocator keeps fragments in distinct registers and waitcnt insertion emits
// partial s_wait_loadcnt/dscnt instead of full drains before every WMMA.
// ---------------------------------------------------------------------------
#if __has_builtin(__builtin_amdgcn_sched_barrier)
#define SCHED_FENCE() __builtin_amdgcn_sched_barrier(0)
#else
#define SCHED_FENCE()
#endif

// ---------------------------------------------------------------------------
// CDNA5 async global->LDS copies (ASYNCcnt-tracked). The gfx1250 builtin takes
// (v4i AS1* gsrc, v4i AS3* ldst, imm offset, imm cpol) per the clang prototype.
// Guarded so the file compiles on toolchains without the builtins.
// ---------------------------------------------------------------------------
#if __has_builtin(__builtin_amdgcn_global_load_async_to_lds_b128) && \
    __has_builtin(__builtin_amdgcn_s_wait_asynccnt)
#define HAVE_ASYNC_LDS 1
typedef __attribute__((address_space(1))) v4i_ gv4i;
typedef __attribute__((address_space(3))) v4i_ lv4i;
#else
#define HAVE_ASYNC_LDS 0
#endif

static __device__ __forceinline__ void copy16_g2l(const __bf16* gsrc, __bf16* ldst) {
#if HAVE_ASYNC_LDS
  __builtin_amdgcn_global_load_async_to_lds_b128(
      (gv4i*)gsrc, (lv4i*)ldst, 0, 0);
#else
  *reinterpret_cast<v8bf*>(ldst) = *reinterpret_cast<const v8bf*>(gsrc);
#endif
}

#if HAVE_ASYNC_LDS
#define ASYNC_WAIT(n) __builtin_amdgcn_s_wait_asynccnt(n)
#else
#define ASYNC_WAIT(n)
#endif

// ---------------------------------------------------------------------------
// gfx1250 WMMA 16x16x32 bf16 fragment loaders (wave32).
// A-matrix 16x32 (MxK): lane row m = lane&15; element i -> k = 16*hf + i
//                       (hf = lane>>4)
// B-matrix 32x16 (KxN): lane col n = lane&15; element i -> k = 16*hf + i
// C/D 16x16 f32: lane col n = lane&15; reg r -> row m = r + 8*hf
// ---------------------------------------------------------------------------
static __device__ __forceinline__ v16bf load_afrag(const __bf16* rowptr, int hf) {
  v8bf lo = *reinterpret_cast<const v8bf*>(rowptr + 16 * hf);       // k = 16hf..16hf+7
  v8bf hi = *reinterpret_cast<const v8bf*>(rowptr + 16 * hf + 8);   // k = 16hf+8..16hf+15
  return __builtin_shufflevector(lo, hi, 0,1,2,3,4,5,6,7,8,9,10,11,12,13,14,15);
}
static __device__ __forceinline__ v16bf load_bfrag(const __bf16* colptr, int hf) {
  // colptr points at B[k=0][n] for this lane's column n, contiguous along k
  return *reinterpret_cast<const v16bf*>(colptr + 16 * hf);
}

// ---------------------------------------------------------------------------
// Conversion / layout kernels
// ---------------------------------------------------------------------------
__global__ void cvt_f32_bf16(const float* __restrict__ src, __bf16* __restrict__ dst, int n) {
  int i = blockIdx.x * blockDim.x + threadIdx.x;
  if (i < n) dst[i] = (__bf16)src[i];
}

// (Hn, Cn, Dn) f32 -> (Hn, Dn, Cn) bf16   (k-contiguous B operand layout)
__global__ void transpose_w(const float* __restrict__ w, __bf16* __restrict__ wt,
                            int Hn, int Cn, int Dn) {
  int i = blockIdx.x * blockDim.x + threadIdx.x;
  int total = Hn * Cn * Dn;
  if (i >= total) return;
  int d = i % Dn;
  int c = (i / Dn) % Cn;
  int hh = i / (Dn * Cn);
  wt[((size_t)hh * Dn + d) * Cn + c] = (__bf16)w[i];
}

// ---------------------------------------------------------------------------
// Fused QKV projection: one wave -> 16(t) x 64(d) tile of q,k,v for one head.
// Per k-step: all 13 fragments load first (SCHED_FENCE), then 12 WMMAs issue
// back-to-back with partial loadcnt waits.
// q,k stored [bh, t, d]; v stored transposed [bh, d, t] (B operand for P@V).
// ---------------------------------------------------------------------------
__global__ void __launch_bounds__(256) qkv_kernel(
    const __bf16* __restrict__ xbf,
    const __bf16* __restrict__ wqT, const __bf16* __restrict__ wkT,
    const __bf16* __restrict__ wvT,
    const float* __restrict__ bq, const float* __restrict__ bk,
    const float* __restrict__ bv,
    __bf16* __restrict__ qw, __bf16* __restrict__ kw, __bf16* __restrict__ vTw) {
  const int lane = threadIdx.x & 31;
  const int hf = lane >> 4;
  const int nl = lane & 15;
  const int wave = blockIdx.x * (blockDim.x >> 5) + (threadIdx.x >> 5);
  const int tiles = TT / 16;                  // 128
  const int bh = wave / tiles;                // 0..63
  const int t0 = (wave % tiles) * 16;
  const int b  = bh / HH;
  const int h  = bh % HH;

  v8f accq[4] = {}, acck[4] = {}, accv[4] = {};
  const __bf16* xrow = xbf + ((size_t)(b * TT + t0 + nl)) * CC;   // A row m = nl

  for (int c0 = 0; c0 < CC; c0 += 32) {
    // ---- issue all loads for this k-step (13 disjoint destinations) ----
    v16bf a = load_afrag(xrow + c0, hf);
    v16bf fq[4], fk[4], fv[4];
#pragma unroll
    for (int j = 0; j < 4; ++j) {
      const size_t off = ((size_t)(h * DHH + 16 * j + nl)) * CC + c0;
      fq[j] = load_bfrag(wqT + off, hf);
      fk[j] = load_bfrag(wkT + off, hf);
      fv[j] = load_bfrag(wvT + off, hf);
    }
    SCHED_FENCE();                            // loads above, WMMAs below
    // ---- 12 back-to-back WMMAs ----
#pragma unroll
    for (int j = 0; j < 4; ++j) {
      accq[j] = __builtin_amdgcn_wmma_f32_16x16x32_bf16(false, a, false, fq[j],
                                                        (short)0, accq[j], false, false);
      acck[j] = __builtin_amdgcn_wmma_f32_16x16x32_bf16(false, a, false, fk[j],
                                                        (short)0, acck[j], false, false);
      accv[j] = __builtin_amdgcn_wmma_f32_16x16x32_bf16(false, a, false, fv[j],
                                                        (short)0, accv[j], false, false);
    }
  }

#pragma unroll
  for (int j = 0; j < 4; ++j) {
    const int dcol = 16 * j + nl;
    const float biq = bq[h * DHH + dcol];
    const float bik = bk[h * DHH + dcol];
    const float biv = bv[h * DHH + dcol];
#pragma unroll
    for (int r = 0; r < 8; ++r) {
      const int t = t0 + r + 8 * hf;
      qw[((size_t)(bh * TT + t)) * DHH + dcol] = (__bf16)(accq[j][r] + biq);
      kw[((size_t)(bh * TT + t)) * DHH + dcol] = (__bf16)(acck[j][r] + bik);
      vTw[((size_t)bh * DHH + dcol) * TT + t]  = (__bf16)(accv[j][r] + biv);
    }
  }
}

// ---------------------------------------------------------------------------
// Block-cooperative causal flash attention.
// One 256-thread block (8 waves) owns 128 consecutive query rows of one (b,h).
// Per 32-key chunk, the block stages K (32x64) and V^T (64x32) tiles into LDS
// with double-buffered async global->LDS copies (ASYNCcnt) so the next chunk's
// DMA overlaps the current chunk's WMMAs + online softmax.
// ---------------------------------------------------------------------------
__global__ void __launch_bounds__(256) attn_kernel(
    const __bf16* __restrict__ qw, const __bf16* __restrict__ kw,
    const __bf16* __restrict__ vTw, __bf16* __restrict__ att) {
  __shared__ __align__(16) __bf16 kbuf[2][32][64];  // [buf][s_local][d]   8 KB
  __shared__ __align__(16) __bf16 vbuf[2][64][32];  // [buf][d][s_local]   8 KB
  __shared__ __align__(16) __bf16 pls[8][16][32];   // per-wave P tiles    8 KB

  const int tid  = threadIdx.x;
  const int lane = tid & 31;
  const int hf = lane >> 4;
  const int nl = lane & 15;
  const int wv = tid >> 5;

  const int blocks_per_bh = TT / 128;                 // 16
  const int bh  = blockIdx.x / blocks_per_bh;
  const int t0b = (blockIdx.x % blocks_per_bh) * 128; // block's first row
  const int t0  = t0b + wv * 16;                      // this wave's first row
  const int b = bh / HH, h = bh % HH;
  const float scale = 0.03125f;                       // C^-0.5 = 1/32

  // Staging thread mapping: each thread moves 16 B of K and 16 B of V per chunk.
  const int kr = tid >> 3, kc = (tid & 7) * 8;        // K tile: 32 rows x 128 B
  const int vr = tid >> 2, vc = (tid & 3) * 8;        // V tile: 64 rows x  64 B
  const __bf16* kg = kw  + ((size_t)bh * TT + kr) * DHH + kc;
  const __bf16* vg = vTw + ((size_t)bh * DHH + vr) * TT + vc;

  // Q fragments (persistent): d = 0..31 and 32..63
  const __bf16* qrow = qw + ((size_t)(bh * TT + t0 + nl)) * DHH;
  v16bf qa0 = load_afrag(qrow, hf);
  v16bf qa1 = load_afrag(qrow + 32, hf);

  v8f o[4] = {};
  float mrow[8], lrow[8];
#pragma unroll
  for (int r = 0; r < 8; ++r) { mrow[r] = -1e30f; lrow[r] = 0.0f; }

  const int nchunks = t0b / 32 + 4;                   // keys 0 .. t0b+127

  // Prefetch chunk 0 (2 async ops per thread)
  copy16_g2l(kg, &kbuf[0][kr][kc]);
  copy16_g2l(vg, &vbuf[0][vr][vc]);

  for (int ci = 0; ci < nchunks; ++ci) {
    const int s0  = ci * 32;
    const int buf = ci & 1;
    if (ci + 1 < nchunks) {                           // prefetch next chunk
      copy16_g2l(kg + (size_t)(ci + 1) * 32 * DHH, &kbuf[buf ^ 1][kr][kc]);
      copy16_g2l(vg + (ci + 1) * 32,               &vbuf[buf ^ 1][vr][vc]);
      ASYNC_WAIT(2);                                  // current chunk landed
    } else {
      ASYNC_WAIT(0);
    }
    __syncthreads();                                  // publish tiles block-wide

    // ---- S = Q K^T over the 32-key chunk (two 16-col subtiles) ----
    v16bf kf[4];
#pragma unroll
    for (int sub = 0; sub < 2; ++sub) {
      const __bf16* kcol = &kbuf[buf][16 * sub + nl][0];
      kf[2 * sub + 0] = *reinterpret_cast<const v16bf*>(kcol + 16 * hf);
      kf[2 * sub + 1] = *reinterpret_cast<const v16bf*>(kcol + 32 + 16 * hf);
    }
    SCHED_FENCE();                                    // ds loads above, WMMAs below
    v8f sc[2] = {};
#pragma unroll
    for (int sub = 0; sub < 2; ++sub) {
      sc[sub] = __builtin_amdgcn_wmma_f32_16x16x32_bf16(false, qa0, false, kf[2 * sub + 0],
                                                        (short)0, sc[sub], false, false);
      sc[sub] = __builtin_amdgcn_wmma_f32_16x16x32_bf16(false, qa1, false, kf[2 * sub + 1],
                                                        (short)0, sc[sub], false, false);
    }
    // ---- scale + causal mask ----
#pragma unroll
    for (int sub = 0; sub < 2; ++sub) {
      const int scol = s0 + 16 * sub + nl;
#pragma unroll
      for (int r = 0; r < 8; ++r) {
        const int trow = t0 + r + 8 * hf;
        const float v = sc[sub][r] * scale;
        sc[sub][r] = (scol <= trow) ? v : -1e30f;
      }
    }
    // ---- online softmax (row r+8*hf lives across the 16 lanes of a half) ----
#pragma unroll
    for (int r = 0; r < 8; ++r) {
      float cm = fmaxf(sc[0][r], sc[1][r]);
      cm = fmaxf(cm, __shfl_xor(cm, 1, 32));
      cm = fmaxf(cm, __shfl_xor(cm, 2, 32));
      cm = fmaxf(cm, __shfl_xor(cm, 4, 32));
      cm = fmaxf(cm, __shfl_xor(cm, 8, 32));
      const float mnew = fmaxf(mrow[r], cm);
      const float corr = __expf(mrow[r] - mnew);
      const float p0 = __expf(sc[0][r] - mnew);
      const float p1 = __expf(sc[1][r] - mnew);
      sc[0][r] = p0;
      sc[1][r] = p1;
      float rs = p0 + p1;
      rs += __shfl_xor(rs, 1, 32);
      rs += __shfl_xor(rs, 2, 32);
      rs += __shfl_xor(rs, 4, 32);
      rs += __shfl_xor(rs, 8, 32);
      lrow[r] = lrow[r] * corr + rs;
      mrow[r] = mnew;
#pragma unroll
      for (int j = 0; j < 4; ++j) o[j][r] *= corr;
    }
    // ---- transpose P: C/D layout -> A layout via per-wave LDS tile ----
#pragma unroll
    for (int sub = 0; sub < 2; ++sub)
#pragma unroll
      for (int r = 0; r < 8; ++r)
        pls[wv][r + 8 * hf][16 * sub + nl] = (__bf16)sc[sub][r];
    asm volatile("s_wait_dscnt 0" ::: "memory");      // wave-private tile: st->ld
    // ---- O += P V : batch P + V fragment loads, then 4 WMMAs ----
    v16bf pa = load_afrag(&pls[wv][nl][0], hf);
    v16bf vf[4];
#pragma unroll
    for (int j = 0; j < 4; ++j)
      vf[j] = *reinterpret_cast<const v16bf*>(&vbuf[buf][16 * j + nl][0] + 16 * hf);
    SCHED_FENCE();                                    // ds loads above, WMMAs below
#pragma unroll
    for (int j = 0; j < 4; ++j)
      o[j] = __builtin_amdgcn_wmma_f32_16x16x32_bf16(false, pa, false, vf[j],
                                                     (short)0, o[j], false, false);
    __syncthreads();                                  // readers done before refill
  }

  // ---- normalize + store concatenated heads (bf16) ----
#pragma unroll
  for (int j = 0; j < 4; ++j) {
#pragma unroll
    for (int r = 0; r < 8; ++r) {
      const int t = t0 + r + 8 * hf;
      att[((size_t)(b * TT + t)) * CC + h * DHH + 16 * j + nl] =
          (__bf16)(o[j][r] / lrow[r]);
    }
  }
}

// ---------------------------------------------------------------------------
// Output projection: out = att @ Wo + bo  (fp32 out).
// One wave -> 16(t) x 64(n) tile, K = 1024. Loads batched before WMMAs.
// ---------------------------------------------------------------------------
__global__ void __launch_bounds__(256) proj_kernel(
    const __bf16* __restrict__ att, const __bf16* __restrict__ woT,
    const float* __restrict__ bo, float* __restrict__ out) {
  const int lane = threadIdx.x & 31;
  const int hf = lane >> 4;
  const int nl = lane & 15;
  const int wave = blockIdx.x * (blockDim.x >> 5) + (threadIdx.x >> 5);
  const int ntiles = CC / 64;                 // 16
  const int rt = wave / ntiles;               // row-tile over B*T
  const int n0 = (wave % ntiles) * 64;
  const int t0 = rt * 16;

  v8f acc[4] = {};
  const __bf16* arow = att + ((size_t)(t0 + nl)) * CC;
  for (int k0 = 0; k0 < CC; k0 += 32) {
    v16bf a = load_afrag(arow + k0, hf);
    v16bf f[4];
#pragma unroll
    for (int j = 0; j < 4; ++j)
      f[j] = load_bfrag(woT + ((size_t)(n0 + 16 * j + nl)) * CC + k0, hf);
    SCHED_FENCE();                            // loads above, WMMAs below
#pragma unroll
    for (int j = 0; j < 4; ++j)
      acc[j] = __builtin_amdgcn_wmma_f32_16x16x32_bf16(false, a, false, f[j],
                                                       (short)0, acc[j], false, false);
  }
#pragma unroll
  for (int j = 0; j < 4; ++j) {
    const float bias = bo[n0 + 16 * j + nl];
#pragma unroll
    for (int r = 0; r < 8; ++r)
      out[((size_t)(t0 + r + 8 * hf)) * CC + n0 + 16 * j + nl] = acc[j][r] + bias;
  }
}

// ---------------------------------------------------------------------------
// Host launch
// ---------------------------------------------------------------------------
extern "C" void kernel_launch(void* const* d_in, const int* in_sizes, int n_in,
                              void* d_out, int out_size, void* d_ws, size_t ws_size,
                              hipStream_t stream) {
  const float* x  = (const float*)d_in[0];
  const float* Wq = (const float*)d_in[1];
  const float* bq = (const float*)d_in[2];
  const float* Wk = (const float*)d_in[3];
  const float* bk = (const float*)d_in[4];
  const float* Wv = (const float*)d_in[5];
  const float* bv = (const float*)d_in[6];
  const float* Wo = (const float*)d_in[7];
  const float* bo = (const float*)d_in[8];
  float* out = (float*)d_out;

  // Workspace layout (bytes); total = 88 MB
  const size_t NX = (size_t)BB * TT * CC;          // 8 Mi elements
  const size_t NW = (size_t)HH * CC * DHH;         // 1 Mi elements
  const size_t NO = (size_t)CC * CC;               // 1 Mi elements
  char* ws = (char*)d_ws;
  __bf16* xbf = (__bf16*)(ws);                     // 16 MB
  __bf16* wqT = (__bf16*)(ws + NX * 2);            //  2 MB
  __bf16* wkT = (__bf16*)(ws + NX * 2 + NW * 2);
  __bf16* wvT = (__bf16*)(ws + NX * 2 + NW * 4);
  __bf16* woT = (__bf16*)(ws + NX * 2 + NW * 6);   //  2 MB
  __bf16* qw  = (__bf16*)(ws + NX * 2 + NW * 6 + NO * 2);              // 16 MB
  __bf16* kw  = (__bf16*)(ws + NX * 2 + NW * 6 + NO * 2 + NX * 2);     // 16 MB
  __bf16* vTw = (__bf16*)(ws + NX * 2 + NW * 6 + NO * 2 + NX * 4);     // 16 MB
  __bf16* atw = (__bf16*)(ws + NX * 2 + NW * 6 + NO * 2 + NX * 6);     // 16 MB

  // 1) precision conversion + weight transposes
  cvt_f32_bf16<<<(int)((NX + 255) / 256), 256, 0, stream>>>(x, xbf, (int)NX);
  transpose_w<<<(int)((NW + 255) / 256), 256, 0, stream>>>(Wq, wqT, HH, CC, DHH);
  transpose_w<<<(int)((NW + 255) / 256), 256, 0, stream>>>(Wk, wkT, HH, CC, DHH);
  transpose_w<<<(int)((NW + 255) / 256), 256, 0, stream>>>(Wv, wvT, HH, CC, DHH);
  transpose_w<<<(int)((NO + 255) / 256), 256, 0, stream>>>(Wo, woT, 1, CC, CC);

  // 2) fused QKV projection: B*H*(T/16) waves, 8 waves/block
  const int qkv_waves = BB * HH * (TT / 16);       // 8192
  qkv_kernel<<<qkv_waves / 8, 256, 0, stream>>>(xbf, wqT, wkT, wvT, bq, bk, bv,
                                                qw, kw, vTw);

  // 3) block-cooperative causal flash attention: one block per 128 rows of (b,h)
  const int attn_blocks = BB * HH * (TT / 128);    // 1024
  attn_kernel<<<attn_blocks, 256, 0, stream>>>(qw, kw, vTw, atw);

  // 4) output projection: (B*T/16)*(C/64) waves
  const int proj_waves = (BB * TT / 16) * (CC / 64);   // 8192
  proj_kernel<<<proj_waves / 8, 256, 0, stream>>>(atw, woT, bo, out);
}
